// SwinPermutatorKanBlock_85031762526278
// MI455X (gfx1250) — matrix operand order repaired and checked by reading, not compiled, not run on hardware
//
#include <hip/hip_runtime.h>
#include <hip/hip_bf16.h>

typedef __attribute__((ext_vector_type(16))) _Float16 v16h;
typedef __attribute__((ext_vector_type(8)))  float    v8f;
typedef __attribute__((ext_vector_type(4)))  float    v4f;

// f16 weight layout inside d_ws (element offsets)
#define WS1_OFF 0
#define WS2_OFF 2048
#define WS3_OFF 4096
#define WS4_OFF 12288
#define WB1_OFF 20480
#define WB2_OFF 20736
#define WB3_OFF 20992
#define WB4_OFF 22016
#define WTOT    23040

__device__ __forceinline__ v8f wmma_f16(v16h a, v16h b, v8f c) {
  // D = A(16x32 f16) * B(32x16 f16) + C(16x16 f32)
  return __builtin_amdgcn_wmma_f32_16x16x32_f16(false, a, false, b, (short)0, c,
                                                false, false);
}

__device__ __forceinline__ float silu_f(float v) {
  // v * sigmoid(v) with fast reciprocal (v_rcp_f32) instead of IEEE divide
  return v * __builtin_amdgcn_rcpf(1.0f + __expf(-v));
}

// Build one A-fragment K-chunk of the spline GEMM: 16 f16 values =
// feature fA (8 RBF grid points) then feature fA+1 (8 RBF grid points).
template <typename GET>
__device__ __forceinline__ v16h make_a_spline(GET get, int t, int fA, float mu,
                                              float rs, const float* lg,
                                              const float* lb) {
  float lnA = (get(t, fA)     - mu) * rs * lg[fA]     + lb[fA];
  float lnB = (get(t, fA + 1) - mu) * rs * lg[fA + 1] + lb[fA + 1];
  v16h a;
#pragma unroll
  for (int g = 0; g < 8; ++g) {
    float grid = -2.0f + (float)g * (4.0f / 7.0f);   // linspace(-2,2,8)
    float dA = (lnA - grid) * 1.75f;                 // /(4/7)
    float dB = (lnB - grid) * 1.75f;
    a[g]     = (_Float16)__expf(-dA * dA);
    a[g + 8] = (_Float16)__expf(-dB * dB);
  }
  return a;
}

// Branch 1/2: tokens=128, in=16, out=16.
// BR1: token=(w,cc), feat=(h,s).  BR2: token=(h,cc), feat=(w,s).
template <bool BR1, typename GET>
__device__ __forceinline__ void run_branch_16(GET get, const _Float16* wsp,
                                              const _Float16* wbp,
                                              const float* smu, const float* srs,
                                              const float* lg, const float* lb,
                                              const float* sbias, float* sZ,
                                              int lane) {
  const int half = lane >> 4;
  const int r    = lane & 15;
  // B fragments: column n = r; lane<16 holds K k0..k0+15, lane>=16 K k0+16..k0+31
  v16h bs[4];
#pragma unroll
  for (int kt = 0; kt < 4; ++kt)
    bs[kt] = *(const v16h*)(wsp + r * 128 + kt * 32 + half * 16);
  v16h bbase = {};
  if (half == 0) bbase = *(const v16h*)(wbp + r * 16);  // K16..31 are zero-pad
  const float bias = sbias[r];

  for (int m0 = 0; m0 < 8; ++m0) {
    const int t = m0 * 16 + r;  // A row = lane%16
    const float mu = smu[t], rs = srs[t];
    v8f acc = {};
#pragma unroll
    for (int kt = 0; kt < 4; ++kt) {
      v16h a = make_a_spline(get, t, kt * 4 + half * 2, mu, rs, lg, lb);
      acc = wmma_f16(a, bs[kt], acc);
    }
    {  // SiLU base path, K=16 zero-padded to 32
      v16h ab = {};
      const int f0 = half * 8;
#pragma unroll
      for (int i = 0; i < 16; ++i) ab[i] = half ? (_Float16)0.0f : (_Float16)silu_f(get(t, f0 + i));
      acc = wmma_f16(ab, bbase, acc);
    }
    // Scatter D (row = v + 8*half, col = r) back into z[h*8+w][cc*2+s]
#pragma unroll
    for (int v = 0; v < 8; ++v) {
      const int trow = m0 * 16 + v + half * 8;
      const int o = r;
      const int hh = BR1 ? (o >> 1)    : (trow >> 4);
      const int ww = BR1 ? (trow >> 4) : (o >> 1);
      const int cc = trow & 15, s = o & 1;
      sZ[(((hh << 3) + ww) << 5) + (cc << 1) + s] += acc[v] + bias;
    }
  }
}

// Branch3 / k4: tokens=64, in=32, out=32.  A fragment is built ONCE per
// (m0,kt) and feeds both N-tiles (two back-to-back WMMAs) — no redundant exps.
template <bool STORE_GLOBAL, typename GET>
__device__ __forceinline__ void run_kan_32(GET get, const _Float16* wsp,
                                           const _Float16* wbp, const float* smu,
                                           const float* srs, const float* lg,
                                           const float* lb, const float* sbias,
                                           float* sZ, float* outp, const float* xin,
                                           int bimg, int row0, int col0, int lane) {
  const int half = lane >> 4;
  const int r    = lane & 15;
  const int n0 = r, n1 = 16 + r;
  const float bias0 = sbias[n0], bias1 = sbias[n1];

  for (int m0 = 0; m0 < 4; ++m0) {
    const int t = m0 * 16 + r;
    const float mu = smu[t], rs = srs[t];
    v8f acc0 = {}, acc1 = {};
#pragma unroll
    for (int kt = 0; kt < 8; ++kt) {
      v16h a  = make_a_spline(get, t, kt * 4 + half * 2, mu, rs, lg, lb);
      v16h b0 = *(const v16h*)(wsp + n0 * 256 + kt * 32 + half * 16);
      v16h b1 = *(const v16h*)(wsp + n1 * 256 + kt * 32 + half * 16);
      acc0 = wmma_f16(a, b0, acc0);
      acc1 = wmma_f16(a, b1, acc1);
    }
    {  // SiLU base path, exact K=32: lane<16 -> f{0..15}, lane>=16 -> f{16..31}
      v16h ab;
      const int f0 = half * 16;
#pragma unroll
      for (int i = 0; i < 8; ++i) {
        ab[i]     = (_Float16)silu_f(get(t, f0 + i));
        ab[i + 8] = (_Float16)silu_f(get(t, f0 + 8 + i));
      }
      v16h bb0 = *(const v16h*)(wbp + n0 * 32 + half * 16);
      v16h bb1 = *(const v16h*)(wbp + n1 * 32 + half * 16);
      acc0 = wmma_f16(ab, bb0, acc0);
      acc1 = wmma_f16(ab, bb1, acc1);
    }
    if (STORE_GLOBAL) {
      // D rows v=0..7 of this (m0,half) are 8 consecutive w pixels of row
      // h = m0*2+half: fully vectorized fused residual + store (b128).
      const int hrow = row0 + m0 * 2 + half;
      size_t g0 = (((size_t)(bimg * 32 + n0) * 256) + hrow) * 256 + col0;
      size_t g1 = (((size_t)(bimg * 32 + n1) * 256) + hrow) * 256 + col0;
      v4f x00 = *(const v4f*)(xin + g0), x01 = *(const v4f*)(xin + g0 + 4);
      v4f x10 = *(const v4f*)(xin + g1), x11 = *(const v4f*)(xin + g1 + 4);
      v4f o00, o01, o10, o11;
#pragma unroll
      for (int j = 0; j < 4; ++j) {
        o00[j] = acc0[j]     + bias0 + x00[j];
        o01[j] = acc0[j + 4] + bias0 + x01[j];
        o10[j] = acc1[j]     + bias1 + x10[j];
        o11[j] = acc1[j + 4] + bias1 + x11[j];
      }
      *(v4f*)(outp + g0)     = o00;
      *(v4f*)(outp + g0 + 4) = o01;
      *(v4f*)(outp + g1)     = o10;
      *(v4f*)(outp + g1 + 4) = o11;
    } else {
#pragma unroll
      for (int v = 0; v < 8; ++v) {
        const int trow = m0 * 16 + v + half * 8;
        sZ[(trow << 5) + n0] += acc0[v] + bias0;
        sZ[(trow << 5) + n1] += acc1[v] + bias1;
      }
    }
  }
}

__global__ __launch_bounds__(32) void swin_kan_kernel(
    const float* __restrict__ x, const _Float16* __restrict__ wf,
    const float* __restrict__ pn_g, const float* __restrict__ pn_b,
    const float* __restrict__ l1g, const float* __restrict__ l1b,
    const float* __restrict__ bb1, const float* __restrict__ l2g,
    const float* __restrict__ l2b, const float* __restrict__ bb2,
    const float* __restrict__ l3g, const float* __restrict__ l3b,
    const float* __restrict__ bb3, const float* __restrict__ l4g,
    const float* __restrict__ l4b, const float* __restrict__ bb4,
    float* __restrict__ out) {
  __shared__ float sY[64 * 32];   // prenorm-LN'd window, token-major
  __shared__ float sZ[64 * 32];   // branch accumulator z = a + d + e
  __shared__ float sMu1[128], sRs1[128], sMu2[128], sRs2[128];
  __shared__ float sMu3[64], sRs3[64], sMu4[64], sRs4[64];
  __shared__ float sPnG[32], sPnB[32];
  __shared__ float sL1g[16], sL1b[16], sL2g[16], sL2b[16];
  __shared__ float sL3g[32], sL3b[32], sL4g[32], sL4b[32];
  __shared__ float sBb1[16], sBb2[16], sBb3[32], sBb4[32];

  const int lane = threadIdx.x;
  const int widx = blockIdx.x;              // 4096 windows
  const int bimg = widx >> 10;
  const int rem  = widx & 1023;
  const int row0 = (rem >> 5) << 3;
  const int col0 = (rem & 31) << 3;

  // ---- load small parameter vectors into LDS ----
  sPnG[lane] = pn_g[lane]; sPnB[lane] = pn_b[lane];
  sL3g[lane] = l3g[lane];  sL3b[lane] = l3b[lane];  sBb3[lane] = bb3[lane];
  sL4g[lane] = l4g[lane];  sL4b[lane] = l4b[lane];  sBb4[lane] = bb4[lane];
  if (lane < 16) {
    sL1g[lane] = l1g[lane]; sL1b[lane] = l1b[lane]; sBb1[lane] = bb1[lane];
    sL2g[lane] = l2g[lane]; sL2b[lane] = l2b[lane]; sBb2[lane] = bb2[lane];
  }

  // ---- phase A: load window as float4 (b c h w -> token-major LDS) ----
#pragma unroll 4
  for (int i = 0; i < 16; ++i) {
    const int idx = i * 32 + lane;          // 512 float4s per window
    const int ch = idx >> 4;
    const int t0 = (idx & 15) << 2;         // 4 consecutive tokens, same row
    size_t gi = (((size_t)(bimg * 32 + ch) * 256) + row0 + (t0 >> 3)) * 256
                + col0 + (t0 & 7);
    v4f vv = *(const v4f*)(x + gi);
#pragma unroll
    for (int j = 0; j < 4; ++j) sY[(t0 + j) * 32 + ch] = vv[j];
  }
  __syncthreads();

  // ---- phase B: prenorm layernorm over C=32, in place ----
  for (int j = 0; j < 2; ++j) {
    const int t = lane + j * 32;
    float sum = 0.f, sq = 0.f;
#pragma unroll
    for (int ch = 0; ch < 32; ++ch) { float v = sY[t * 32 + ch]; sum += v; sq += v * v; }
    const float mu = sum * (1.f / 32.f);
    const float rs = rsqrtf(fmaxf(sq * (1.f / 32.f) - mu * mu, 0.f) + 1e-5f);
#pragma unroll
    for (int ch = 0; ch < 32; ++ch)
      sY[t * 32 + ch] = (sY[t * 32 + ch] - mu) * rs * sPnG[ch] + sPnB[ch];
  }
  __syncthreads();

  // ---- branch-input getters (permutations of y) ----
  auto getB1 = [&](int t, int f) -> float {  // t=(w,cc) f=(h,s)
    int w = t >> 4, cc = t & 15, h = f >> 1, s = f & 1;
    return sY[(((h << 3) + w) << 5) + (cc << 1) + s];
  };
  auto getB2 = [&](int t, int f) -> float {  // t=(h,cc) f=(w,s)
    int h = t >> 4, cc = t & 15, w = f >> 1, s = f & 1;
    return sY[(((h << 3) + w) << 5) + (cc << 1) + s];
  };
  auto getB3 = [&](int t, int f) -> float { return sY[(t << 5) + f]; };
  auto getB4 = [&](int t, int f) -> float { return sZ[(t << 5) + f]; };

  // ---- phase C: per-branch fastkan-LN stats; zero z ----
  for (int j = 0; j < 4; ++j) {
    const int t = lane + j * 32;
    float s1 = 0.f, q1 = 0.f, s2 = 0.f, q2 = 0.f;
#pragma unroll
    for (int f = 0; f < 16; ++f) {
      float v1 = getB1(t, f); s1 += v1; q1 += v1 * v1;
      float v2 = getB2(t, f); s2 += v2; q2 += v2 * v2;
    }
    float mu = s1 * (1.f / 16.f);
    sMu1[t] = mu; sRs1[t] = rsqrtf(fmaxf(q1 * (1.f / 16.f) - mu * mu, 0.f) + 1e-5f);
    mu = s2 * (1.f / 16.f);
    sMu2[t] = mu; sRs2[t] = rsqrtf(fmaxf(q2 * (1.f / 16.f) - mu * mu, 0.f) + 1e-5f);
  }
  for (int j = 0; j < 2; ++j) {
    const int t = lane + j * 32;
    float s3 = 0.f, q3 = 0.f;
#pragma unroll
    for (int f = 0; f < 32; ++f) { float v = sY[t * 32 + f]; s3 += v; q3 += v * v; }
    const float mu = s3 * (1.f / 32.f);
    sMu3[t] = mu; sRs3[t] = rsqrtf(fmaxf(q3 * (1.f / 32.f) - mu * mu, 0.f) + 1e-5f);
  }
  for (int i = lane; i < 2048; i += 32) sZ[i] = 0.f;
  __syncthreads();

  // ---- phase D: three branches (WMMA) accumulating into z ----
  run_branch_16<true >(getB1, wf + WS1_OFF, wf + WB1_OFF, sMu1, sRs1, sL1g, sL1b, sBb1, sZ, lane);
  __syncthreads();
  run_branch_16<false>(getB2, wf + WS2_OFF, wf + WB2_OFF, sMu2, sRs2, sL2g, sL2b, sBb2, sZ, lane);
  __syncthreads();
  run_kan_32<false>(getB3, wf + WS3_OFF, wf + WB3_OFF, sMu3, sRs3, sL3g, sL3b,
                    sBb3, sZ, out, x, bimg, row0, col0, lane);
  __syncthreads();

  // ---- phase E: k4 LN stats over z ----
  for (int j = 0; j < 2; ++j) {
    const int t = lane + j * 32;
    float s4 = 0.f, q4 = 0.f;
#pragma unroll
    for (int f = 0; f < 32; ++f) { float v = sZ[t * 32 + f]; s4 += v; q4 += v * v; }
    const float mu = s4 * (1.f / 32.f);
    sMu4[t] = mu; sRs4[t] = rsqrtf(fmaxf(q4 * (1.f / 32.f) - mu * mu, 0.f) + 1e-5f);
  }
  __syncthreads();

  // ---- phase F: k4 (WMMA) + fused residual + window-reverse b128 store ----
  run_kan_32<true>(getB4, wf + WS4_OFF, wf + WB4_OFF, sMu4, sRs4, sL4g, sL4b,
                   sBb4, sZ, out, x, bimg, row0, col0, lane);
}

// Convert all KAN weights to f16 into workspace.
__global__ void prep_weights(const float* ws1, const float* wb1, const float* ws2,
                             const float* wb2, const float* ws3, const float* wb3,
                             const float* ws4, const float* wb4, _Float16* wf) {
  const int i = blockIdx.x * blockDim.x + threadIdx.x;
  if (i < 2048)       wf[WS1_OFF + i]           = (_Float16)ws1[i];
  else if (i < 4096)  wf[WS2_OFF + (i - 2048)]  = (_Float16)ws2[i - 2048];
  else if (i < 12288) wf[WS3_OFF + (i - 4096)]  = (_Float16)ws3[i - 4096];
  else if (i < 20480) wf[WS4_OFF + (i - 12288)] = (_Float16)ws4[i - 12288];
  else if (i < 20736) wf[WB1_OFF + (i - 20480)] = (_Float16)wb1[i - 20480];
  else if (i < 20992) wf[WB2_OFF + (i - 20736)] = (_Float16)wb2[i - 20736];
  else if (i < 22016) wf[WB3_OFF + (i - 20992)] = (_Float16)wb3[i - 20992];
  else if (i < WTOT)  wf[WB4_OFF + (i - 22016)] = (_Float16)wb4[i - 22016];
}

extern "C" void kernel_launch(void* const* d_in, const int* in_sizes, int n_in,
                              void* d_out, int out_size, void* d_ws, size_t ws_size,
                              hipStream_t stream) {
  (void)in_sizes; (void)n_in; (void)out_size; (void)ws_size;
  const float* x    = (const float*)d_in[0];
  const float* pn_g = (const float*)d_in[1];
  const float* pn_b = (const float*)d_in[2];
  const float* l1g = (const float*)d_in[3],  *l1b = (const float*)d_in[4];
  const float* ws1 = (const float*)d_in[5],  *wb1 = (const float*)d_in[6];
  const float* bb1 = (const float*)d_in[7];
  const float* l2g = (const float*)d_in[8],  *l2b = (const float*)d_in[9];
  const float* ws2 = (const float*)d_in[10], *wb2 = (const float*)d_in[11];
  const float* bb2 = (const float*)d_in[12];
  const float* l3g = (const float*)d_in[13], *l3b = (const float*)d_in[14];
  const float* ws3 = (const float*)d_in[15], *wb3 = (const float*)d_in[16];
  const float* bb3 = (const float*)d_in[17];
  const float* l4g = (const float*)d_in[18], *l4b = (const float*)d_in[19];
  const float* ws4 = (const float*)d_in[20], *wb4 = (const float*)d_in[21];
  const float* bb4 = (const float*)d_in[22];

  _Float16* wf = (_Float16*)d_ws;
  prep_weights<<<(WTOT + 255) / 256, 256, 0, stream>>>(ws1, wb1, ws2, wb2, ws3,
                                                       wb3, ws4, wb4, wf);
  swin_kan_kernel<<<4096, 32, 0, stream>>>(x, wf, pn_g, pn_b, l1g, l1b, bb1, l2g,
                                           l2b, bb2, l3g, l3b, bb3, l4g, l4b, bb4,
                                           (float*)d_out);
}
